// MoE_31516470018497
// MI455X (gfx1250) — compile-verified, matrix operand order checked by reading, not run on hardware
//
#include <hip/hip_runtime.h>
#include <hip/hip_bf16.h>

// Problem constants (from reference setup_inputs)
#define BT    2048      // B*T tokens
#define HD    1024      // hidden
#define FFD   1024      // ffn inner
#define NE    16        // experts
#define NK    4         // top-k
#define SLIM  7.0f      // swiglu clip limit

typedef float v2f  __attribute__((ext_vector_type(2)));
typedef float v8f  __attribute__((ext_vector_type(8)));
typedef __bf16 v16bf __attribute__((ext_vector_type(16)));

// Workspace layout (bytes)
#define OFF_CNT   0
#define OFF_IMP   64
#define OFF_LOAD  128
#define OFF_TOK   256
#define OFF_WL    (OFF_TOK + NE * BT * 4)            // 131328
#define OFF_W1S   (OFF_WL + NE * BT * 4)             // 262400 (32B aligned)
#define OFF_W2S   (OFF_W1S + (size_t)NE * HD * 2 * FFD * 2)  // +67108864
#define WS_NEED   (OFF_W2S + (size_t)NE * FFD * HD * 2)      // +33554432

static __device__ __forceinline__ unsigned short f2bf(float f) {
  unsigned u = __float_as_uint(f);
  unsigned r = u + 0x7fffu + ((u >> 16) & 1u);   // round-to-nearest-even
  return (unsigned short)(r >> 16);
}

// ---------------------------------------------------------------------------
// Router: one block per token. logits -> softmax -> importance/load atomics,
// top-4 (first-index tie-break), normalized weights, per-expert token lists.
// ---------------------------------------------------------------------------
__global__ __launch_bounds__(64) void moe_router_kernel(
    const float* __restrict__ x, const float* __restrict__ rw,
    const float* __restrict__ rb, int* __restrict__ cnt,
    float* __restrict__ imp, float* __restrict__ loadv,
    int* __restrict__ tok_list, float* __restrict__ w_list) {
  const int t   = blockIdx.x;
  const int tid = threadIdx.x;
  __shared__ float red[NE * 64];
  __shared__ float logits[NE];

  float acc[NE];
#pragma unroll
  for (int e = 0; e < NE; ++e) acc[e] = 0.0f;
  for (int h = tid; h < HD; h += 64) {
    const float xv = x[(size_t)t * HD + h];
#pragma unroll
    for (int e = 0; e < NE; ++e) acc[e] += xv * rw[(size_t)e * HD + h];
  }
#pragma unroll
  for (int e = 0; e < NE; ++e) red[e * 64 + tid] = acc[e];
  __syncthreads();

  if (tid < NE) {
    float s = 0.0f;
    for (int i = 0; i < 64; ++i) s += red[tid * 64 + i];
    logits[tid] = s + rb[tid];
  }
  __syncthreads();

  if (tid == 0) {
    float p[NE];
    float m = -1e30f;
    for (int e = 0; e < NE; ++e) m = fmaxf(m, logits[e]);
    float s = 0.0f;
    for (int e = 0; e < NE; ++e) { p[e] = __expf(logits[e] - m); s += p[e]; }
    const float inv = 1.0f / s;
    for (int e = 0; e < NE; ++e) p[e] *= inv;

    for (int e = 0; e < NE; ++e) atomicAdd(&imp[e], p[e]);
    int a1 = 0;
    for (int e = 1; e < NE; ++e) if (p[e] > p[a1]) a1 = e;
    atomicAdd(&loadv[a1], 1.0f);

    bool used[NE];
    for (int e = 0; e < NE; ++e) used[e] = false;
    float wv[NK]; int id[NK]; float wsum = 0.0f;
    for (int k = 0; k < NK; ++k) {
      int best = 0; float bv = -1e30f;
      for (int e = 0; e < NE; ++e)
        if (!used[e] && p[e] > bv) { bv = p[e]; best = e; }
      used[best] = true; id[k] = best; wv[k] = bv; wsum += bv;
    }
    const float invw = 1.0f / (wsum + 1e-9f);
    for (int k = 0; k < NK; ++k) {
      const int e   = id[k];
      const int pos = atomicAdd(&cnt[e], 1);
      tok_list[e * BT + pos] = t;
      w_list[e * BT + pos]   = wv[k] * invw;
    }
  }
}

// ---------------------------------------------------------------------------
__global__ void moe_aux_kernel(const float* __restrict__ imp,
                               const float* __restrict__ loadv,
                               float* __restrict__ out_aux) {
  if (threadIdx.x == 0 && blockIdx.x == 0) {
    const float invBT = 1.0f / (float)BT;
    float s = 0.0f;
    for (int e = 0; e < NE; ++e) s += (imp[e] * invBT) * (loadv[e] * invBT);
    *out_aux = (float)NE * s;
  }
}

// ---------------------------------------------------------------------------
// Weight swizzle+convert: f32 row-major [E][K=1024][N] -> bf16 tiles in the
// exact wave32 WMMA 16x16x32 B-fragment lane order.
// Tile = (kc: 32 K) x (nt: 16 N), 512 bf16, laid out lane-major, 16 bf16/lane:
//   lane = h*16 + n%16  (h = K half select)
//   per-lane bf16 jj: VGPR j = jj/2 holds K = jgrp*16 + h*8 + jr*2 + parity
// Each block: one (e,kc) slab of 8 n-tiles (32x128 f32) staged via LDS so
// both global reads (512B rows) and swizzled writes (8KB run) are coalesced.
// ---------------------------------------------------------------------------
__global__ __launch_bounds__(256) void moe_swizzle_kernel(
    const float* __restrict__ W, unsigned int* __restrict__ dst,
    int Ncols /* 2048 or 1024 */) {
  __shared__ float tile[32 * 128];
  const int gc  = Ncols / 128;                 // n-groups per (e,kc)
  const int bx  = blockIdx.x;
  const int e   = bx / (32 * gc);
  const int rem = bx % (32 * gc);
  const int kc  = rem / gc;
  const int g   = rem % gc;
  const int tid = threadIdx.x;

  for (int i = tid; i < 32 * 128; i += 256) {
    const int kr = i >> 7;
    const int c  = i & 127;
    tile[i] = W[((size_t)e * HD + (kc * 32 + kr)) * Ncols + g * 128 + c];
  }
  __syncthreads();

  const size_t baseWord = ((size_t)(e * 32 + kc) * (Ncols / 16) + g * 8) * 256;
  for (int w = tid; w < 2048; w += 256) {
    const int elem0 = 2 * w;
    const int tl    = elem0 >> 9;        // local n-tile 0..7
    const int within = elem0 & 511;
    const int lane  = within >> 4;
    const int jj0   = within & 15;       // even
    const int h     = lane >> 4;
    const int nlo   = lane & 15;
    const int j     = jj0 >> 1;
    const int jgrp  = j >> 2;
    const int jr    = j & 3;
    const int k0r   = jgrp * 16 + h * 8 + jr * 2;
    const int nloc  = tl * 16 + nlo;
    const float s0  = tile[k0r * 128 + nloc];
    const float s1  = tile[(k0r + 1) * 128 + nloc];
    dst[baseWord + w] = (unsigned int)f2bf(s0) |
                        ((unsigned int)f2bf(s1) << 16);
  }
}

// ---------------------------------------------------------------------------
// Fused expert FFN, bf16 WMMA (16x16x32, fp32 accumulate). Block = (expert,
// 64-token tile), 8 waves. Phase 1: up/gate GEMM + SwiGLU -> act bf16 in LDS.
// Phase 2: down GEMM + weighted combine via f32 global atomics.
// ---------------------------------------------------------------------------
__global__ __launch_bounds__(256) void moe_ffn_bf16_kernel(
    const float* __restrict__ x,
    const unsigned int* __restrict__ w1s, const float* __restrict__ b1,
    const unsigned int* __restrict__ w2s, const float* __restrict__ b2,
    const int* __restrict__ cnt, const int* __restrict__ tok_list,
    const float* __restrict__ w_list, float* __restrict__ out) {
  __shared__ unsigned short Xs[64 * HD];    // 128 KB, bf16 X tile
  __shared__ unsigned short As[64 * FFD];   // 128 KB, bf16 act tile

  const int e     = blockIdx.y;
  const int n     = cnt[e];
  const int tile0 = blockIdx.x * 64;
  if (tile0 >= n) return;                   // block-uniform; EXEC stays ones

  const int tid  = threadIdx.x;
  const int lane = tid & 31;
  const int wave = tid >> 5;                // 0..7
  const int h    = lane >> 4;               // K-half select
  const int nlo  = lane & 15;               // A-row / B-col within tile

  const int*   toks = tok_list + e * BT + tile0;
  const float* wts  = w_list   + e * BT + tile0;

  // Gather 64 token rows of x, convert f32->bf16, pack into LDS.
  unsigned int* Xw = (unsigned int*)Xs;
  for (int i = tid; i < 64 * (HD / 2); i += 256) {
    const int row = i >> 9;
    const int cp  = i & 511;
    unsigned int v = 0;
    if (tile0 + row < n) {
      const float2 f = *(const float2*)&x[(size_t)toks[row] * HD + cp * 2];
      v = (unsigned int)f2bf(f.x) | ((unsigned int)f2bf(f.y) << 16);
    }
    Xw[i] = v;
  }
  __syncthreads();

  const v16bf* W1v = (const v16bf*)w1s + (size_t)(e * 32) * 128 * 32;
  const v16bf* W2v = (const v16bf*)w2s + (size_t)(e * 32) * 64 * 32;
  const uint4* Xs4 = (const uint4*)Xs;
  const uint4* As4 = (const uint4*)As;

  union AFrag { uint4 q[2]; v16bf v; };

  // ---------------- Phase 1: up/gate GEMM + SwiGLU -------------------------
#pragma unroll
  for (int i = 0; i < 8; ++i) {
    const int nt = wave * 8 + i;            // up n-tile (0..63)
    const int f0 = nt * 16;
    v8f cu[4], cg[4];
#pragma unroll
    for (int m = 0; m < 4; ++m) { cu[m] = {}; cg[m] = {}; }

    for (int kc = 0; kc < 32; ++kc) {
      const v16bf bu = W1v[((size_t)kc * 128 + nt) * 32 + lane];
      const v16bf bg = W1v[((size_t)kc * 128 + 64 + nt) * 32 + lane];
      AFrag a[4];
#pragma unroll
      for (int m = 0; m < 4; ++m) {
        const int base = (m * 16 + nlo) * 128 + kc * 4 + h;
        a[m].q[0] = Xs4[base];
        a[m].q[1] = Xs4[base + 2];
      }
#pragma unroll
      for (int m = 0; m < 4; ++m) {
        cu[m] = __builtin_amdgcn_wmma_f32_16x16x32_bf16(
            false, a[m].v, false, bu, (short)0, cu[m], false, false);
        cg[m] = __builtin_amdgcn_wmma_f32_16x16x32_bf16(
            false, a[m].v, false, bg, (short)0, cg[m], false, false);
      }
    }
    const float biasu = b1[e * (2 * FFD) + f0 + nlo];
    const float biasg = b1[e * (2 * FFD) + FFD + f0 + nlo];
#pragma unroll
    for (int m = 0; m < 4; ++m) {
#pragma unroll
      for (int r = 0; r < 8; ++r) {
        float up = cu[m][r] + biasu;
        float g  = cg[m][r] + biasg;
        up = fminf(fmaxf(up, -SLIM), SLIM);
        g  = fminf(fmaxf(g,  -SLIM), SLIM);
        const float sig = 1.0f / (1.0f + __expf(-g));
        const int Mrow = m * 16 + r + 8 * h;
        As[Mrow * FFD + f0 + nlo] = f2bf((g * sig) * up);
      }
    }
  }
  __syncthreads();

  // ---------------- Phase 2: down GEMM + weighted combine ------------------
#pragma unroll
  for (int i = 0; i < 8; ++i) {
    const int nt = wave * 8 + i;            // h n-tile (0..63)
    const int h0 = nt * 16;
    v8f cz[4];
#pragma unroll
    for (int m = 0; m < 4; ++m) cz[m] = {};

    for (int kc = 0; kc < 32; ++kc) {
      const v16bf b = W2v[((size_t)kc * 64 + nt) * 32 + lane];
      AFrag a[4];
#pragma unroll
      for (int m = 0; m < 4; ++m) {
        const int base = (m * 16 + nlo) * 128 + kc * 4 + h;
        a[m].q[0] = As4[base];
        a[m].q[1] = As4[base + 2];
      }
#pragma unroll
      for (int m = 0; m < 4; ++m)
        cz[m] = __builtin_amdgcn_wmma_f32_16x16x32_bf16(
            false, a[m].v, false, b, (short)0, cz[m], false, false);
    }
    const float bias2 = b2[e * HD + h0 + nlo];
#pragma unroll
    for (int m = 0; m < 4; ++m) {
#pragma unroll
      for (int r = 0; r < 8; ++r) {
        const int Mrow = m * 16 + r + 8 * h;
        if (tile0 + Mrow < n) {
          atomicAdd(&out[(size_t)toks[Mrow] * HD + h0 + nlo],
                    wts[Mrow] * (cz[m][r] + bias2));
        }
      }
    }
  }
}

// ---------------------------------------------------------------------------
// Fallback fp32-WMMA FFN (16-token tiles) when workspace can't hold bf16
// weights. Same math, V_WMMA_F32_16X16X4_F32.
// ---------------------------------------------------------------------------
__global__ __launch_bounds__(256) void moe_ffn_f32_kernel(
    const float* __restrict__ x,  const float* __restrict__ w1,
    const float* __restrict__ b1, const float* __restrict__ w2,
    const float* __restrict__ b2, const int* __restrict__ cnt,
    const int* __restrict__ tok_list, const float* __restrict__ w_list,
    float* __restrict__ out) {
  __shared__ float lds[16 * HD];

  const int e     = blockIdx.y;
  const int n     = cnt[e];
  const int tile0 = blockIdx.x * 16;
  if (tile0 >= n) return;

  const int tid  = threadIdx.x;
  const int lane = tid & 31;
  const int wave = tid >> 5;

  const int*   toks = tok_list + e * BT + tile0;
  const float* wts  = w_list   + e * BT + tile0;

  for (int i = tid; i < 16 * HD; i += 256) {
    const int row = i >> 10;
    const int col = i & (HD - 1);
    float v = 0.0f;
    if (tile0 + row < n) v = x[(size_t)toks[row] * HD + col];
    lds[i] = v;
  }
  __syncthreads();

  const float* W1 = w1 + (size_t)e * HD * (2 * FFD);
  const float* W2 = w2 + (size_t)e * FFD * HD;
  const int colN  = lane & 15;
  const int khalf = (lane >> 4) * 2;
  const int rowM  = lane & 15;
  const int mBase = (lane >> 4) * 8;

  v8f actreg[8];
#pragma unroll
  for (int i = 0; i < 8; ++i) {
    const int f0 = (wave * 8 + i) * 16;
    v8f cu = {}; v8f cg = {};
    const float* Bu = W1 + f0 + colN;
    const float* Bg = W1 + FFD + f0 + colN;
#pragma unroll 2
    for (int k0 = 0; k0 < HD; k0 += 4) {
      const int kr = k0 + khalf;
      v2f a; const float2 av = *(const float2*)&lds[rowM * HD + kr];
      a.x = av.x; a.y = av.y;
      v2f bu, bg;
      bu.x = Bu[(size_t)kr * (2 * FFD)]; bu.y = Bu[(size_t)(kr + 1) * (2 * FFD)];
      bg.x = Bg[(size_t)kr * (2 * FFD)]; bg.y = Bg[(size_t)(kr + 1) * (2 * FFD)];
      cu = __builtin_amdgcn_wmma_f32_16x16x4_f32(false, a, false, bu, (short)0, cu, false, false);
      cg = __builtin_amdgcn_wmma_f32_16x16x4_f32(false, a, false, bg, (short)0, cg, false, false);
    }
    const float biasu = b1[e * (2 * FFD) + f0 + colN];
    const float biasg = b1[e * (2 * FFD) + FFD + f0 + colN];
#pragma unroll
    for (int r = 0; r < 8; ++r) {
      float up = cu[r] + biasu;
      float g  = cg[r] + biasg;
      up = fminf(fmaxf(up, -SLIM), SLIM);
      g  = fminf(fmaxf(g,  -SLIM), SLIM);
      const float sig = 1.0f / (1.0f + __expf(-g));
      actreg[i][r] = (g * sig) * up;
    }
  }
  __syncthreads();
#pragma unroll
  for (int i = 0; i < 8; ++i) {
    const int f0 = (wave * 8 + i) * 16;
#pragma unroll
    for (int r = 0; r < 8; ++r)
      lds[(mBase + r) * HD + f0 + colN] = actreg[i][r];
  }
  __syncthreads();
#pragma unroll
  for (int i = 0; i < 8; ++i) {
    const int h0 = (wave * 8 + i) * 16;
    v8f cz = {};
    const float* B2 = W2 + h0 + colN;
#pragma unroll 2
    for (int k0 = 0; k0 < FFD; k0 += 4) {
      const int kr = k0 + khalf;
      v2f a; const float2 av = *(const float2*)&lds[rowM * HD + kr];
      a.x = av.x; a.y = av.y;
      v2f b;
      b.x = B2[(size_t)kr * HD]; b.y = B2[(size_t)(kr + 1) * HD];
      cz = __builtin_amdgcn_wmma_f32_16x16x4_f32(false, a, false, b, (short)0, cz, false, false);
    }
    const float bias2 = b2[e * HD + h0 + colN];
#pragma unroll
    for (int r = 0; r < 8; ++r) {
      const int M = mBase + r;
      if (tile0 + M < n)
        atomicAdd(&out[(size_t)toks[M] * HD + h0 + colN],
                  wts[M] * (cz[r] + bias2));
    }
  }
}

// ---------------------------------------------------------------------------
extern "C" void kernel_launch(void* const* d_in, const int* in_sizes, int n_in,
                              void* d_out, int out_size, void* d_ws,
                              size_t ws_size, hipStream_t stream) {
  (void)in_sizes; (void)n_in;
  const float* x  = (const float*)d_in[0];
  const float* w1 = (const float*)d_in[1];
  const float* b1 = (const float*)d_in[2];
  const float* w2 = (const float*)d_in[3];
  const float* b2 = (const float*)d_in[4];
  const float* rw = (const float*)d_in[5];
  const float* rb = (const float*)d_in[6];
  float* out = (float*)d_out;

  char*  ws       = (char*)d_ws;
  int*   cnt      = (int*)(ws + OFF_CNT);
  float* imp      = (float*)(ws + OFF_IMP);
  float* loadv    = (float*)(ws + OFF_LOAD);
  int*   tok_list = (int*)(ws + OFF_TOK);
  float* w_list   = (float*)(ws + OFF_WL);
  unsigned int* w1s = (unsigned int*)(ws + OFF_W1S);
  unsigned int* w2s = (unsigned int*)(ws + OFF_W2S);

  hipMemsetAsync(d_out, 0, (size_t)out_size * sizeof(float), stream);
  hipMemsetAsync(d_ws, 0, 256, stream);

  moe_router_kernel<<<BT, 64, 0, stream>>>(x, rw, rb, cnt, imp, loadv,
                                           tok_list, w_list);
  moe_aux_kernel<<<1, 32, 0, stream>>>(imp, loadv, out + (out_size - 1));

  if (ws_size >= WS_NEED) {
    // bf16 WMMA path: swizzle-convert weights, then 64-token-tile FFN.
    moe_swizzle_kernel<<<NE * 32 * (2 * FFD / 128), 256, 0, stream>>>(
        w1, w1s, 2 * FFD);
    moe_swizzle_kernel<<<NE * 32 * (HD / 128), 256, 0, stream>>>(
        w2, w2s, HD);
    moe_ffn_bf16_kernel<<<dim3(BT / 64, NE), 256, 0, stream>>>(
        x, w1s, b1, w2s, b2, cnt, tok_list, w_list, out);
  } else {
    moe_ffn_f32_kernel<<<dim3(BT / 16, NE), 256, 0, stream>>>(
        x, w1, b1, w2, b2, cnt, tok_list, w_list, out);
  }
}
